// PerlinAttention_73598559584999
// MI455X (gfx1250) — matrix operand order repaired and checked by reading, not compiled, not run on hardware
//
#include <hip/hip_runtime.h>
#include <math.h>

// ---------------------------------------------------------------------------
// Perlin attention forward for gfx1250 (MI455X).
// GEMM stages: v_wmma_f32_16x16x32_f16, one wave computes 16xMB*16 tiles.
// All K/N dims padded so inner loops have zero lane-divergent guards.
// CNN stages: direct f32 VALU kernels (L2-resident, trivial vs WMMA peak).
// ---------------------------------------------------------------------------

typedef __attribute__((ext_vector_type(16))) _Float16 v16h;
typedef __attribute__((ext_vector_type(8)))  float    v8f;

#define NB    8
#define NH    12
#define TT    512
#define HID   64
#define LL    128
#define NF    266          // performer features
#define FP    288          // NF padded to multiple of 32 (covers a_kmap reach)
#define NT    (NB*NH*TT)   // 49152 rows
#define NHEAD (NB*NH)      // 96

#define C1    48
#define OH1   255
#define OW1   63
#define OHT   511
#define OWT   127
#define OH2   509
#define OW2   125

// A-matrix (16x32 f16) k index for vector element e, half hi = lane>>4
__device__ __forceinline__ int a_kmap(int e, int hi) {
    return e + hi * 16;
}
__device__ __forceinline__ float wred_sum(float v) {
    for (int o = 16; o > 0; o >>= 1) v += __shfl_xor(v, o, 32);
    return v;
}
__device__ __forceinline__ float wred_max(float v) {
    for (int o = 16; o > 0; o >>= 1) v = fmaxf(v, __shfl_xor(v, o, 32));
    return v;
}
__device__ __forceinline__ void atomicMaxF(float* addr, float val) {
    unsigned int* ua = (unsigned int*)addr;
    unsigned int cur = *ua;
    while (__uint_as_float(cur) < val) {
        unsigned int prev = atomicCAS(ua, cur, __float_as_uint(val));
        if (prev == cur) break;
        cur = prev;
    }
}
__device__ __forceinline__ int iclamp(int v, int lo, int hi) {
    return v < lo ? lo : (v > hi ? hi : v);
}
__device__ __forceinline__ int refl(int i, int n) {
    if (i < 0) return -i;
    if (i >= n) return 2 * n - 2 - i;
    return i;
}

// ---------------------------------------------------------------------------
// K0a: vmask identity rows (grid_sample of eye), mask flags, init global max
// ---------------------------------------------------------------------------
__global__ void k_eye(const float* __restrict__ mask, float* __restrict__ eye,
                      float* __restrict__ flag, float* __restrict__ gmax) {
    int n = blockIdx.x;
    int t = threadIdx.x;            // 512 threads
    if (n == 0 && t == 0) *gmax = -INFINITY;
    const float* mrow = mask + n * TT;
    float zsum = 0.f, zcum = 0.f;
    for (int i = 0; i < TT; ++i) {
        float z = (mrow[i] > -1.f) ? 1.f : 0.f;
        zsum += z;
        if (i <= t) zcum += z;
    }
    float ty  = ((zcum - 1.f) / (zsum - 1.f + 1e-8f)) * 2.f - 1.f;
    float pix = (ty + 1.f) * 0.5f * (float)(HID - 1);
    float y0f = floorf(pix);
    float fy  = pix - y0f;
    int   y0  = (int)y0f;
    flag[n * TT + t] = (mrow[t] < -1.f) ? 0.f : 1.f;
    float* er = eye + (n * TT + t) * HID;
    for (int e = 0; e < HID; ++e) {
        float v = 0.f;
        if (e == y0)     v += 1.f - fy;
        if (e == y0 + 1) v += fy;
        er[e] = v;
    }
}

// ---------------------------------------------------------------------------
// K0b: projp[f,d] = (f<NF ? proj[f,d] : 0) * dn    (288x64, fold dn scale)
// ---------------------------------------------------------------------------
__global__ void k_padproj(const float* __restrict__ proj, float* __restrict__ projp) {
    int idx = blockIdx.x * 256 + threadIdx.x;
    if (idx >= FP * HID) return;
    int f = idx / HID;
    projp[idx] = (f < NF) ? proj[idx] * 0.35355339059327373f : 0.f;
}

// ---------------------------------------------------------------------------
// K0c: materialize v_fa = flag * [eye | v_for_atten]  (NT x 128)
// ---------------------------------------------------------------------------
__global__ void k_vfa(const float* __restrict__ eye, const float* __restrict__ vfa_in,
                      const float* __restrict__ flag, float* __restrict__ vfab) {
    int idx = blockIdx.x * 256 + threadIdx.x;
    if (idx >= NT * LL) return;
    int e = idx % LL;
    int row = idx / LL;           // head*TT + t
    int head = row / TT;
    int t = row - head * TT;
    int nb = head / NH;
    float fl = flag[nb * TT + t];
    float val = (e < HID) ? eye[(nb * TT + t) * HID + e]
                          : vfa_in[row * HID + (e - HID)];
    vfab[idx] = val * fl;
}

// ---------------------------------------------------------------------------
// K1: dd = data @ (dn*proj)^T for q_for_atten (z=0) / k_for_atten (z=1)
//     4 M-tiles per wave; also global max of dd_k (valid cols only)
// ---------------------------------------------------------------------------
#define MB_DD 4
__global__ void k_dd(const float* __restrict__ qa, const float* __restrict__ ka,
                     const float* __restrict__ projp,
                     float* __restrict__ ddq, float* __restrict__ ddk,
                     float* __restrict__ gmax) {
    const int lane = threadIdx.x, lo = lane & 15, hi = lane >> 4;
    const int ntile = blockIdx.x;           // 0..17
    const int mg = blockIdx.y;              // 0..767
    const int which = blockIdx.z;
    const float* src = which ? ka : qa;
    float* dst = which ? ddk : ddq;
    const int n = ntile * 16 + lo;
    const v8f z = {0.f, 0.f, 0.f, 0.f, 0.f, 0.f, 0.f, 0.f};
    v8f acc[MB_DD];
#pragma unroll
    for (int mb = 0; mb < MB_DD; ++mb) acc[mb] = z;
#pragma unroll
    for (int kb = 0; kb < HID; kb += 32) {
        v16h b;
#pragma unroll
        for (int e = 0; e < 16; ++e)
            b[e] = (_Float16)projp[n * HID + kb + hi * 16 + e];
#pragma unroll
        for (int mb = 0; mb < MB_DD; ++mb) {
            const int m = (mg * MB_DD + mb) * 16 + lo;
            v16h a;
#pragma unroll
            for (int e = 0; e < 16; ++e)
                a[e] = (_Float16)src[m * HID + kb + a_kmap(e, hi)];
            acc[mb] = __builtin_amdgcn_wmma_f32_16x16x32_f16(
                false, a, false, b, (short)0, acc[mb], false, false);
        }
    }
    float tmax = -INFINITY;
#pragma unroll
    for (int mb = 0; mb < MB_DD; ++mb)
#pragma unroll
        for (int r = 0; r < 8; ++r) {
            int mm = (mg * MB_DD + mb) * 16 + hi * 8 + r;
            dst[mm * FP + n] = acc[mb][r];
            if (which && n < NF) tmax = fmaxf(tmax, acc[mb][r]);
        }
    if (which) {
        tmax = wred_max(tmax);
        if (lane == 0) atomicMaxF(gmax, tmax);
    }
}

// ---------------------------------------------------------------------------
// K2: phi transform in place: ratio * (exp(dd - diag - m) + eps)
//     q rows use per-row max, k rows use global max; pads (f>=266) stay 0.
// ---------------------------------------------------------------------------
__global__ void k_phi(const float* __restrict__ qa, const float* __restrict__ ka,
                      float* __restrict__ ddq, float* __restrict__ ddk,
                      const float* __restrict__ gmax) {
    int wid = threadIdx.x >> 5, lane = threadIdx.x & 31;
    int id = blockIdx.x * 8 + wid;           // 0 .. 2*NT-1
    int which = id / NT;
    int row = id - which * NT;
    const float* src = which ? ka : qa;
    float* buf = which ? ddk : ddq;
    float s = 0.f;
    for (int j = lane; j < HID; j += 32) { float x = src[row * HID + j]; s += x * x; }
    s = wred_sum(s);
    float diag = s * 0.5f * 0.125f;          // 0.5 * dn^2
    float m;
    if (which) {
        m = *gmax;
    } else {
        float mx = -INFINITY;
        for (int f = lane; f < NF; f += 32) mx = fmaxf(mx, buf[row * FP + f]);
        m = wred_max(mx);
    }
    const float ratio = 0.06131393394849658f;  // 266^-0.5
    for (int f = lane; f < NF; f += 32) {
        float dd = buf[row * FP + f];
        buf[row * FP + f] = ratio * (__expf(dd - diag - m) + 1e-4f);
    }
}

// ---------------------------------------------------------------------------
// K3: kp_sum[head,f] = sum_t kp[head,t,f]
// ---------------------------------------------------------------------------
__global__ void k_kpsum(const float* __restrict__ kp, float* __restrict__ kpsum) {
    int idx = blockIdx.x * blockDim.x + threadIdx.x;
    if (idx >= NHEAD * FP) return;
    int head = idx / FP, f = idx - head * FP;
    float s = 0.f;
    for (int t = 0; t < TT; ++t) s += kp[(head * TT + t) * FP + f];
    kpsum[idx] = s;
}

// ---------------------------------------------------------------------------
// K4: ctx[head] = kp^T (FxT) @ v_fa (TxE); 2 M-tiles per wave
// ---------------------------------------------------------------------------
#define MB_CTX 2
__global__ void k_ctx(const float* __restrict__ kp, const float* __restrict__ vfab,
                      float* __restrict__ ctx) {
    const int lane = threadIdx.x, lo = lane & 15, hi = lane >> 4;
    const int ntile = blockIdx.x;       // 0..7  (E cols)
    const int mg = blockIdx.y;          // 0..8  (F rows /2 tiles)
    const int head = blockIdx.z;
    const int n = ntile * 16 + lo;
    const v8f z = {0.f, 0.f, 0.f, 0.f, 0.f, 0.f, 0.f, 0.f};
    v8f acc[MB_CTX];
#pragma unroll
    for (int mb = 0; mb < MB_CTX; ++mb) acc[mb] = z;
    for (int kb = 0; kb < TT; kb += 32) {
        v16h b;
#pragma unroll
        for (int e = 0; e < 16; ++e) {
            int t = kb + hi * 16 + e;
            b[e] = (_Float16)vfab[(head * TT + t) * LL + n];
        }
#pragma unroll
        for (int mb = 0; mb < MB_CTX; ++mb) {
            const int m = (mg * MB_CTX + mb) * 16 + lo;
            v16h a;
#pragma unroll
            for (int e = 0; e < 16; ++e) {
                int t = kb + a_kmap(e, hi);
                a[e] = (_Float16)kp[(head * TT + t) * FP + m];
            }
            acc[mb] = __builtin_amdgcn_wmma_f32_16x16x32_f16(
                false, a, false, b, (short)0, acc[mb], false, false);
        }
    }
#pragma unroll
    for (int mb = 0; mb < MB_CTX; ++mb)
#pragma unroll
        for (int r = 0; r < 8; ++r) {
            int mm = (mg * MB_CTX + mb) * 16 + hi * 8 + r;
            ctx[(head * FP + mm) * LL + n] = acc[mb][r];
        }
}

// ---------------------------------------------------------------------------
// K5: d_inv[row] = 1 / (qp[row,:] . kp_sum[head,:])
// ---------------------------------------------------------------------------
__global__ void k_dinv(const float* __restrict__ qp, const float* __restrict__ kpsum,
                       float* __restrict__ dinv) {
    int row = blockIdx.x * blockDim.x + threadIdx.x;
    if (row >= NT) return;
    int head = row / TT;
    float s = 0.f;
    for (int f = 0; f < FP; ++f) s += qp[row * FP + f] * kpsum[head * FP + f];
    dinv[row] = 1.f / s;
}

// ---------------------------------------------------------------------------
// K6: pcl[head] = (qp (TxF) @ ctx (FxE)) * d_inv; 4 M-tiles per wave
// ---------------------------------------------------------------------------
#define MB_PCL 4
__global__ void k_pcl(const float* __restrict__ qp, const float* __restrict__ ctx,
                      const float* __restrict__ dinv, float* __restrict__ pcl) {
    const int lane = threadIdx.x, lo = lane & 15, hi = lane >> 4;
    const int ntile = blockIdx.x;       // 0..7
    const int mg = blockIdx.y;          // 0..7
    const int head = blockIdx.z;
    const int n = ntile * 16 + lo;
    const v8f z = {0.f, 0.f, 0.f, 0.f, 0.f, 0.f, 0.f, 0.f};
    v8f acc[MB_PCL];
#pragma unroll
    for (int mb = 0; mb < MB_PCL; ++mb) acc[mb] = z;
    for (int kb = 0; kb < FP; kb += 32) {        // 9 full chunks, no guards
        v16h b;
#pragma unroll
        for (int e = 0; e < 16; ++e)
            b[e] = (_Float16)ctx[(head * FP + kb + hi * 16 + e) * LL + n];
#pragma unroll
        for (int mb = 0; mb < MB_PCL; ++mb) {
            const int m = (mg * MB_PCL + mb) * 16 + lo;
            v16h a;
#pragma unroll
            for (int e = 0; e < 16; ++e)
                a[e] = (_Float16)qp[(head * TT + m) * FP + kb + a_kmap(e, hi)];
            acc[mb] = __builtin_amdgcn_wmma_f32_16x16x32_f16(
                false, a, false, b, (short)0, acc[mb], false, false);
        }
    }
#pragma unroll
    for (int mb = 0; mb < MB_PCL; ++mb)
#pragma unroll
        for (int r = 0; r < 8; ++r) {
            int mm = (mg * MB_PCL + mb) * 16 + hi * 8 + r;
            pcl[(head * TT + mm) * LL + n] = acc[mb][r] * dinv[head * TT + mm];
        }
}

// ---------------------------------------------------------------------------
// K7: enc = [pcl | v_m] (NTx192) @ W_enc (192x128) + b_enc; split-K phases
// ---------------------------------------------------------------------------
#define MB_ENC 4
__global__ void k_enc(const float* __restrict__ pcl, const float* __restrict__ v,
                      const float* __restrict__ flag,
                      const float* __restrict__ Wenc, const float* __restrict__ benc,
                      float* __restrict__ enc) {
    const int lane = threadIdx.x, lo = lane & 15, hi = lane >> 4;
    const int ntile = blockIdx.x;       // 0..7
    const int mg = blockIdx.y;          // 0..767
    const int n = ntile * 16 + lo;
    const v8f z = {0.f, 0.f, 0.f, 0.f, 0.f, 0.f, 0.f, 0.f};
    v8f acc[MB_ENC];
    float fl[MB_ENC];
#pragma unroll
    for (int mb = 0; mb < MB_ENC; ++mb) {
        acc[mb] = z;
        int m = (mg * MB_ENC + mb) * 16 + lo;
        int nb = m / (NH * TT);
        int t = m % TT;
        fl[mb] = flag[nb * TT + t];
    }
    // phase 1: K = 0..127 from pcl
#pragma unroll
    for (int kb = 0; kb < LL; kb += 32) {
        v16h b;
#pragma unroll
        for (int e = 0; e < 16; ++e)
            b[e] = (_Float16)Wenc[(kb + hi * 16 + e) * LL + n];
#pragma unroll
        for (int mb = 0; mb < MB_ENC; ++mb) {
            const int m = (mg * MB_ENC + mb) * 16 + lo;
            v16h a;
#pragma unroll
            for (int e = 0; e < 16; ++e)
                a[e] = (_Float16)pcl[m * LL + kb + a_kmap(e, hi)];
            acc[mb] = __builtin_amdgcn_wmma_f32_16x16x32_f16(
                false, a, false, b, (short)0, acc[mb], false, false);
        }
    }
    // phase 2: K = 128..191 from v * flag
#pragma unroll
    for (int kb = 0; kb < HID; kb += 32) {
        v16h b;
#pragma unroll
        for (int e = 0; e < 16; ++e)
            b[e] = (_Float16)Wenc[(LL + kb + hi * 16 + e) * LL + n];
#pragma unroll
        for (int mb = 0; mb < MB_ENC; ++mb) {
            const int m = (mg * MB_ENC + mb) * 16 + lo;
            v16h a;
#pragma unroll
            for (int e = 0; e < 16; ++e)
                a[e] = (_Float16)(v[m * HID + kb + a_kmap(e, hi)] * fl[mb]);
            acc[mb] = __builtin_amdgcn_wmma_f32_16x16x32_f16(
                false, a, false, b, (short)0, acc[mb], false, false);
        }
    }
#pragma unroll
    for (int mb = 0; mb < MB_ENC; ++mb)
#pragma unroll
        for (int r = 0; r < 8; ++r) {
            int mm = (mg * MB_ENC + mb) * 16 + hi * 8 + r;
            enc[mm * LL + n] = acc[mb][r] + benc[n];
        }
}

// ---------------------------------------------------------------------------
// K8: LayerNorm(128) + exact GELU, in place; one wave per row
// ---------------------------------------------------------------------------
__global__ void k_ln(float* __restrict__ enc, const float* __restrict__ g,
                     const float* __restrict__ b) {
    int wid = threadIdx.x >> 5, lane = threadIdx.x & 31;
    int row = blockIdx.x * 8 + wid;
    float x[4];
    float s = 0.f;
#pragma unroll
    for (int j = 0; j < 4; ++j) { x[j] = enc[row * LL + lane + 32 * j]; s += x[j]; }
    s = wred_sum(s);
    float mu = s * (1.f / LL);
    float var = 0.f;
#pragma unroll
    for (int j = 0; j < 4; ++j) { float d = x[j] - mu; var += d * d; }
    var = wred_sum(var) * (1.f / LL);
    float rs = rsqrtf(var + 1e-5f);
#pragma unroll
    for (int j = 0; j < 4; ++j) {
        int c = lane + 32 * j;
        float y = (x[j] - mu) * rs * g[c] + b[c];
        enc[row * LL + c] = 0.5f * y * (1.f + erff(y * 0.70710678118654752f));
    }
}

// ---------------------------------------------------------------------------
// K9: score = tpred (NTx128) @ W_dec (128x128) + b_dec; 4 M-tiles per wave
// ---------------------------------------------------------------------------
#define MB_DEC 4
__global__ void k_dec(const float* __restrict__ tp, const float* __restrict__ Wdec,
                      const float* __restrict__ bdec, float* __restrict__ score) {
    const int lane = threadIdx.x, lo = lane & 15, hi = lane >> 4;
    const int ntile = blockIdx.x;
    const int mg = blockIdx.y;
    const int n = ntile * 16 + lo;
    const v8f z = {0.f, 0.f, 0.f, 0.f, 0.f, 0.f, 0.f, 0.f};
    v8f acc[MB_DEC];
#pragma unroll
    for (int mb = 0; mb < MB_DEC; ++mb) acc[mb] = z;
#pragma unroll
    for (int kb = 0; kb < LL; kb += 32) {
        v16h b;
#pragma unroll
        for (int e = 0; e < 16; ++e)
            b[e] = (_Float16)Wdec[(kb + hi * 16 + e) * LL + n];
#pragma unroll
        for (int mb = 0; mb < MB_DEC; ++mb) {
            const int m = (mg * MB_DEC + mb) * 16 + lo;
            v16h a;
#pragma unroll
            for (int e = 0; e < 16; ++e)
                a[e] = (_Float16)tp[m * LL + kb + a_kmap(e, hi)];
            acc[mb] = __builtin_amdgcn_wmma_f32_16x16x32_f16(
                false, a, false, b, (short)0, acc[mb], false, false);
        }
    }
#pragma unroll
    for (int mb = 0; mb < MB_DEC; ++mb)
#pragma unroll
        for (int r = 0; r < 8; ++r) {
            int mm = (mg * MB_DEC + mb) * 16 + hi * 8 + r;
            score[mm * LL + n] = acc[mb][r] + bdec[n];
        }
}

// ---------------------------------------------------------------------------
// CNN stages (direct f32 convs; inputs L2-resident)
// ---------------------------------------------------------------------------
__global__ void k_conv1(const float* __restrict__ in, const float* __restrict__ w,
                        const float* __restrict__ b, float* __restrict__ out) {
    int idx = blockIdx.x * 256 + threadIdx.x;
    if (idx >= NB * C1 * OH1 * OW1) return;
    int x = idx % OW1; int t = idx / OW1;
    int y = t % OH1;  t /= OH1;
    int co = t % C1;  int n = t / C1;
    float s = b[co];
    for (int ci = 0; ci < NH; ++ci) {
        const float* ip = in + ((n * NH + ci) * TT + 2 * y) * LL + 2 * x;
        const float* wp = w + (co * NH + ci) * 9;
#pragma unroll
        for (int dy = 0; dy < 3; ++dy)
#pragma unroll
            for (int dx = 0; dx < 3; ++dx)
                s += ip[dy * LL + dx] * wp[dy * 3 + dx];
    }
    out[idx] = fmaxf(s, 0.f);
}

__global__ void k_rconv(const float* __restrict__ in, const float* __restrict__ skip,
                        const float* __restrict__ w, const float* __restrict__ b,
                        float* __restrict__ out) {
    int idx = blockIdx.x * 256 + threadIdx.x;
    if (idx >= NB * C1 * OH1 * OW1) return;
    int x = idx % OW1; int t = idx / OW1;
    int y = t % OH1;  t /= OH1;
    int co = t % C1;  int n = t / C1;
    int yy0 = refl(y - 1, OH1), yy1 = y, yy2 = refl(y + 1, OH1);
    int xx0 = refl(x - 1, OW1), xx1 = x, xx2 = refl(x + 1, OW1);
    int yr[3] = {yy0, yy1, yy2};
    int xr[3] = {xx0, xx1, xx2};
    float s = b[co];
    for (int ci = 0; ci < C1; ++ci) {
        const float* ip = in + ((n * C1 + ci) * OH1) * OW1;
        const float* wp = w + (co * C1 + ci) * 9;
#pragma unroll
        for (int dy = 0; dy < 3; ++dy)
#pragma unroll
            for (int dx = 0; dx < 3; ++dx)
                s += ip[yr[dy] * OW1 + xr[dx]] * wp[dy * 3 + dx];
    }
    if (skip) s += skip[idx];
    out[idx] = fmaxf(s, 0.f);
}

__global__ void k_convt(const float* __restrict__ in, const float* __restrict__ w,
                        const float* __restrict__ b, float* __restrict__ out) {
    int idx = blockIdx.x * 256 + threadIdx.x;
    if (idx >= NB * NH * OHT * OWT) return;
    int x = idx % OWT; int t = idx / OWT;
    int y = t % OHT;  t /= OHT;
    int co = t % NH;  int n = t / NH;
    float s = b[co];
    for (int ky = 0; ky < 3; ++ky) {
        int ty = y - ky;
        if (ty < 0 || (ty & 1)) continue;
        int iy = ty >> 1;
        if (iy >= OH1) continue;
        for (int kx = 0; kx < 3; ++kx) {
            int tx = x - kx;
            if (tx < 0 || (tx & 1)) continue;
            int ix = tx >> 1;
            if (ix >= OW1) continue;
            for (int ci = 0; ci < C1; ++ci)
                s += in[((n * C1 + ci) * OH1 + iy) * OW1 + ix] *
                     w[((ci * NH + co) * 3 + ky) * 3 + kx];
        }
    }
    out[idx] = fmaxf(s, 0.f);
}

__global__ void k_conv2(const float* __restrict__ in, const float* __restrict__ w,
                        const float* __restrict__ b, float* __restrict__ out) {
    int idx = blockIdx.x * 256 + threadIdx.x;
    if (idx >= NB * NH * OH2 * OW2) return;
    int x = idx % OW2; int t = idx / OW2;
    int y = t % OH2;  t /= OH2;
    int co = t % NH;  int n = t / NH;
    float s = b[co];
    for (int ci = 0; ci < NH; ++ci) {
        const float* ip = in + ((n * NH + ci) * OHT + y) * OWT + x;
        const float* wp = w + (co * NH + ci) * 9;
#pragma unroll
        for (int dy = 0; dy < 3; ++dy)
#pragma unroll
            for (int dx = 0; dx < 3; ++dx)
                s += ip[dy * OWT + dx] * wp[dy * 3 + dx];
    }
    out[idx] = s;
}

__global__ void k_resize(const float* __restrict__ in, float* __restrict__ out) {
    int idx = blockIdx.x * 256 + threadIdx.x;
    if (idx >= NB * NH * TT * LL) return;
    int x = idx % LL; int t = idx / LL;
    int y = t % TT;  t /= TT;
    int c = t % NH;  int n = t / NH;
    float cy = (y + 0.5f) * ((float)OH2 / (float)TT) - 0.5f;
    float cx = (x + 0.5f) * ((float)OW2 / (float)LL) - 0.5f;
    int y0 = (int)floorf(cy); float fy = cy - (float)y0;
    int x0 = (int)floorf(cx); float fx = cx - (float)x0;
    int ya = iclamp(y0, 0, OH2 - 1), yb = iclamp(y0 + 1, 0, OH2 - 1);
    int xa = iclamp(x0, 0, OW2 - 1), xb = iclamp(x0 + 1, 0, OW2 - 1);
    const float* p = in + (n * NH + c) * OH2 * OW2;
    float v = (1.f - fy) * ((1.f - fx) * p[ya * OW2 + xa] + fx * p[ya * OW2 + xb])
            +        fy  * ((1.f - fx) * p[yb * OW2 + xa] + fx * p[yb * OW2 + xb]);
    out[idx] = v;
}

// ---------------------------------------------------------------------------
// Orchestration
// ---------------------------------------------------------------------------
extern "C" void kernel_launch(void* const* d_in, const int* in_sizes, int n_in,
                              void* d_out, int out_size, void* d_ws, size_t ws_size,
                              hipStream_t stream) {
    const float* v      = (const float*)d_in[2];
    const float* qa     = (const float*)d_in[3];
    const float* ka     = (const float*)d_in[4];
    const float* vfa_in = (const float*)d_in[5];
    const float* mask   = (const float*)d_in[8];
    const float* proj   = (const float*)d_in[11];
    const float* Wenc   = (const float*)d_in[12];
    const float* benc   = (const float*)d_in[13];
    const float* lng    = (const float*)d_in[14];
    const float* lnb    = (const float*)d_in[15];
    const float* Wdec   = (const float*)d_in[16];
    const float* bdec   = (const float*)d_in[17];
    const float* c1w    = (const float*)d_in[18];
    const float* c1b    = (const float*)d_in[19];
    const float* r1w1   = (const float*)d_in[20];
    const float* r1b1   = (const float*)d_in[21];
    const float* r1w2   = (const float*)d_in[22];
    const float* r1b2   = (const float*)d_in[23];
    const float* r2w1   = (const float*)d_in[24];
    const float* r2b1   = (const float*)d_in[25];
    const float* r2w2   = (const float*)d_in[26];
    const float* r2b2   = (const float*)d_in[27];
    const float* ctw    = (const float*)d_in[28];
    const float* ctb    = (const float*)d_in[29];
    const float* c2w    = (const float*)d_in[30];
    const float* c2b    = (const float*)d_in[31];

    float* ws    = (float*)d_ws;
    float* eye   = ws;                        //   262,144
    float* flag  = eye + 262144;              //     4,096
    float* gmax  = flag + 4096;               //        64
    float* kpsum = gmax + 64;                 //    27,648
    float* dinv  = kpsum + 27648;             //    49,152
    float* projp = dinv + 49152;              //    18,432
    float* R1    = projp + 18432;             // 14,155,776 (qp -> score, ct)
    float* R2    = R1 + 14155776;             // 14,155,776 (kp -> c1, c2)
    float* ctxb  = R2 + 14155776;             //  3,538,944
    float* pclb  = ctxb + 3538944;            //  6,291,456 (-> rbB)
    float* encb  = pclb + 6291456;            //  6,291,456 (vfab -> enc -> rbC)
    // total 44,794,944 floats = 171 MB

    float* qp    = R1;
    float* kp    = R2;
    float* vfab  = encb;               // dead before k_enc writes enc
    float* score = R1;                 // qp dead after k_pcl
    float* ct    = R1 + 6291456;       // score dead after k_conv1
    float* c1    = R2;                 // kp dead after k_ctx/k_kpsum
    float* c2    = R2 + 6291456;
    float* rbB   = pclb;               // pcl dead after k_enc
    float* rbC   = encb;               // enc dead after k_dec

    // Phase 0: preparation
    k_eye<<<NB, TT, 0, stream>>>(mask, eye, flag, gmax);
    k_padproj<<<(FP * HID + 255) / 256, 256, 0, stream>>>(proj, projp);
    k_vfa<<<(NT * LL + 255) / 256, 256, 0, stream>>>(eye, vfa_in, flag, vfab);

    // Phase 1: performer features
    k_dd<<<dim3(FP / 16, NT / 16 / MB_DD, 2), 32, 0, stream>>>(qa, ka, projp, qp, kp, gmax);
    k_phi<<<(2 * NT) / 8, 256, 0, stream>>>(qa, ka, qp, kp, gmax);
    k_kpsum<<<(NHEAD * FP + 255) / 256, 256, 0, stream>>>(kp, kpsum);

    // Phase 2: linear attention
    k_ctx<<<dim3(LL / 16, (FP / 16) / MB_CTX, NHEAD), 32, 0, stream>>>(kp, vfab, ctxb);
    k_dinv<<<(NT + 255) / 256, 256, 0, stream>>>(qp, kpsum, dinv);
    k_pcl<<<dim3(LL / 16, (TT / 16) / MB_PCL, NHEAD), 32, 0, stream>>>(qp, ctxb, dinv, pclb);

    // Phase 3: MLP
    k_enc<<<dim3(LL / 16, NT / 16 / MB_ENC), 32, 0, stream>>>(pclb, v, flag, Wenc, benc, encb);
    k_ln<<<NT / 8, 256, 0, stream>>>(encb, lng, lnb);
    k_dec<<<dim3(LL / 16, NT / 16 / MB_DEC), 32, 0, stream>>>(encb, Wdec, bdec, score);

    // Phase 4: CNN
    const int t1 = NB * C1 * OH1 * OW1;
    k_conv1<<<(t1 + 255) / 256, 256, 0, stream>>>(score, c1w, c1b, c1);
    k_rconv<<<(t1 + 255) / 256, 256, 0, stream>>>(c1, nullptr, r1w1, r1b1, rbB);
    k_rconv<<<(t1 + 255) / 256, 256, 0, stream>>>(rbB, c1, r1w2, r1b2, rbC);
    k_rconv<<<(t1 + 255) / 256, 256, 0, stream>>>(rbC, nullptr, r2w1, r2b1, rbB);
    k_rconv<<<(t1 + 255) / 256, 256, 0, stream>>>(rbB, rbC, r2w2, r2b2, c1);
    const int tt = NB * NH * OHT * OWT;
    k_convt<<<(tt + 255) / 256, 256, 0, stream>>>(c1, ctw, ctb, ct);
    const int t2 = NB * NH * OH2 * OW2;
    k_conv2<<<(t2 + 255) / 256, 256, 0, stream>>>(ct, c2w, c2b, c2);
    const int to = NB * NH * TT * LL;
    k_resize<<<(to + 255) / 256, 256, 0, stream>>>(c2, (float*)d_out);

    (void)in_sizes; (void)n_in; (void)out_size; (void)ws_size;
}
